// H2GT_HGNN_11235634446345
// MI455X (gfx1250) — matrix operand order repaired and checked by reading, not compiled, not run on hardware
//
#include <hip/hip_runtime.h>
#include <hip/hip_bf16.h>
#include <stdint.h>

// ---------------------------------------------------------------------------
// Problem constants (match reference)
// ---------------------------------------------------------------------------
#define NNODES 8192
#define FIN    512
#define HIDF   256
#define OUTF   128
#define DATT   64

typedef __bf16 bf16;
typedef __attribute__((ext_vector_type(16))) __bf16 v16bf;
typedef __attribute__((ext_vector_type(8)))  __bf16 v8bf;
typedef __attribute__((ext_vector_type(8)))  float  v8f;

// ---------------------------------------------------------------------------
// WMMA fragment loaders (contiguous k order k = 16h + s, not the ISA table order; wave32)
//   A 16x32 bf16 : lanes 0-15 -> M=0..15, K = k0+0..15
//                  lanes 16-31 -> M=0..15, K = k0+16..31
//   B 32x16 bf16 : lanes 0-15 -> K=k0+0..15, lanes 16-31 -> K=k0+16..31,
//                  N = lane&15  (so B stored transposed [N][K] is contiguous)
// ---------------------------------------------------------------------------
__device__ inline v16bf load_a_frag(const bf16* __restrict__ A, size_t lda,
                                    int m, int k0, int laneHi) {
  const bf16* p = A + (size_t)m * lda + (size_t)k0 + 16 * laneHi;
  v8bf lo = *(const v8bf*)p;         // 16B aligned
  v8bf hi = *(const v8bf*)(p + 8);   // 16B aligned
  v16bf r;
#pragma unroll
  for (int i = 0; i < 8; ++i) { r[i] = lo[i]; r[i + 8] = hi[i]; }
  return r;
}

__device__ inline v16bf load_a_frag(const float* __restrict__ A, size_t lda,
                                    int m, int k0, int laneHi) {
  const float* p = A + (size_t)m * lda + (size_t)k0 + 16 * laneHi;
  v8f lo = *(const v8f*)p;           // 32B aligned
  v8f hi = *(const v8f*)(p + 8);     // 32B aligned
  v16bf r;
#pragma unroll
  for (int i = 0; i < 8; ++i) { r[i] = (bf16)lo[i]; r[i + 8] = (bf16)hi[i]; }
  return r;
}

// ---------------------------------------------------------------------------
// Tiled bf16 WMMA GEMM:  Out[M,N] = A[M,K] * B[K,N] (+bias) (relu?)
//   A: row-major (f32 or bf16, converted in-register)
//   B: pre-transposed bf16  BT[N][K]
//   Each wave computes a 16 x 128 strip: 8 accumulation tiles reuse one A frag
//   OMODE: 0 = f32 row-major, 1 = bf16 row-major, 2 = bf16 transposed [N][M]
// ---------------------------------------------------------------------------
template <typename TA, int OMODE, bool BIAS, bool RELU>
__global__ __launch_bounds__(256) void gemm_wmma_kernel(
    const TA* __restrict__ A, const bf16* __restrict__ BT,
    const float* __restrict__ bias, void* __restrict__ Out,
    int M, int N, int K) {
  int tid    = blockIdx.x * 256 + threadIdx.x;
  int wid    = tid >> 5;
  int lane   = threadIdx.x & 31;
  int mtiles = M >> 4;
  int nstrips = N >> 7;
  if (wid >= mtiles * nstrips) return;  // wave-uniform: EXEC stays all-ones

  int mt = wid % mtiles;
  int ns = wid / mtiles;
  int m0 = mt << 4;
  int n0 = ns << 7;
  int laneHi = lane >> 4;
  int laneLo = lane & 15;
  int mA = m0 + laneLo;  // both lane halves load the same 16 rows, K halves

  v8f acc[8];
#pragma unroll
  for (int t = 0; t < 8; ++t)
#pragma unroll
    for (int r = 0; r < 8; ++r) acc[t][r] = 0.0f;

  for (int k0 = 0; k0 < K; k0 += 32) {
    v16bf a = load_a_frag(A, (size_t)K, mA, k0, laneHi);
    const bf16* bbase = BT + (size_t)(n0 + laneLo) * K + k0 + 16 * laneHi;
#pragma unroll
    for (int t = 0; t < 8; ++t) {
      v16bf b = *(const v16bf*)(bbase + (size_t)t * 16 * K);  // 32B load
      acc[t] = __builtin_amdgcn_wmma_f32_16x16x32_bf16(
          false, a, false, b, (short)0, acc[t], false, false);
    }
  }

  // Epilogue: C/D tile layout — VGPR r holds (M = m0+r+8*laneHi, N = n0+t*16+laneLo)
#pragma unroll
  for (int t = 0; t < 8; ++t) {
    int n = n0 + t * 16 + laneLo;
    float bv = BIAS ? bias[n] : 0.0f;
#pragma unroll
    for (int r = 0; r < 8; ++r) {
      int m = m0 + r + 8 * laneHi;
      float v = acc[t][r] + bv;
      if (RELU) v = v > 0.0f ? v : 0.0f;
      if (OMODE == 0)       ((float*)Out)[(size_t)m * N + n] = v;
      else if (OMODE == 1)  ((bf16*)Out)[(size_t)m * N + n] = (bf16)v;
      else                  ((bf16*)Out)[(size_t)n * M + m] = (bf16)v;
    }
  }
}

template <typename TA, int OMODE, bool BIAS, bool RELU>
static void launch_gemm(const TA* A, const bf16* BT, const float* bias,
                        void* Out, int M, int N, int K, hipStream_t s) {
  int waves  = (M / 16) * (N / 128);
  int blocks = (waves * 32 + 255) / 256;
  gemm_wmma_kernel<TA, OMODE, BIAS, RELU><<<blocks, 256, 0, s>>>(
      A, BT, bias, Out, M, N, K);
}

// ---------------------------------------------------------------------------
// Small helper kernels
// ---------------------------------------------------------------------------

// W[K][N] f32 -> WT[N][K] bf16
__global__ void convT_bf16_kernel(const float* __restrict__ W,
                                  bf16* __restrict__ WT, int K, int Nn) {
  int idx = blockIdx.x * blockDim.x + threadIdx.x;
  if (idx >= K * Nn) return;
  int k = idx / Nn, n = idx % Nn;
  WT[(size_t)n * K + k] = (bf16)W[(size_t)k * Nn + n];
}

__global__ void fill_f32_kernel(float* __restrict__ p, float v, int n) {
  int i = blockIdx.x * blockDim.x + threadIdx.x;
  if (i < n) p[i] = v;
}

__global__ void relu_inplace_kernel(float* __restrict__ p, int n) {
  int i = blockIdx.x * blockDim.x + threadIdx.x;
  if (i < n) { float v = p[i]; p[i] = v > 0.0f ? v : 0.0f; }
}

// zs = z @ a_src ; zd = z @ a_dst   (z: [N,128] f32)
__global__ void vecmat_kernel(const float* __restrict__ z,
                              const float* __restrict__ a_src,
                              const float* __restrict__ a_dst,
                              float* __restrict__ zs, float* __restrict__ zd,
                              int n) {
  int r = blockIdx.x * blockDim.x + threadIdx.x;
  if (r >= n) return;
  float s = 0.0f, t = 0.0f;
  const float* row = z + (size_t)r * OUTF;
#pragma unroll 4
  for (int k = 0; k < OUTF; ++k) { float v = row[k]; s += v * a_src[k]; t += v * a_dst[k]; }
  zs[r] = s; zd[r] = t;
}

__device__ inline float leaky02(float x) { return x > 0.0f ? x : 0.2f * x; }

__device__ inline void atomicMaxF(float* addr, float val) {
  if (val >= 0.0f) atomicMax((int*)addr, __float_as_int(val));
  else             atomicMin((unsigned int*)addr, __float_as_uint(val));
}

// pass 1: segment max of leaky(e) over destination
__global__ void edge_max_kernel(const int* __restrict__ src,
                                const int* __restrict__ dst,
                                const float* __restrict__ zs,
                                const float* __restrict__ zd,
                                float* __restrict__ mmax, int E, int n) {
  int i = blockIdx.x * blockDim.x + threadIdx.x;
  int total = E + n;
  if (i >= total) return;
  int s = (i < E) ? src[i] : (i - E);
  int d = (i < E) ? dst[i] : (i - E);
  atomicMaxF(&mmax[d], leaky02(zs[s] + zd[d]));
}

// pass 2: segment sum of exp(e - m[d])
__global__ void edge_sum_kernel(const int* __restrict__ src,
                                const int* __restrict__ dst,
                                const float* __restrict__ zs,
                                const float* __restrict__ zd,
                                const float* __restrict__ mmax,
                                float* __restrict__ denom, int E, int n) {
  int i = blockIdx.x * blockDim.x + threadIdx.x;
  int total = E + n;
  if (i >= total) return;
  int s = (i < E) ? src[i] : (i - E);
  int d = (i < E) ? dst[i] : (i - E);
  float e = leaky02(zs[s] + zd[d]);
  atomicAdd(&denom[d], __expf(e - mmax[d]));
}

// pass 3: out[d,:] += alpha * z[s,:]  — one block (128 thr) per edge
__global__ void edge_scatter_kernel(const int* __restrict__ src,
                                    const int* __restrict__ dst,
                                    const float* __restrict__ zs,
                                    const float* __restrict__ zd,
                                    const float* __restrict__ mmax,
                                    const float* __restrict__ denom,
                                    const float* __restrict__ z,
                                    float* __restrict__ outf, int E, int n) {
  int i = blockIdx.x;
  int c = threadIdx.x;
  int s = (i < E) ? src[i] : (i - E);
  int d = (i < E) ? dst[i] : (i - E);
  float e = leaky02(zs[s] + zd[d]);
  float alpha = __expf(e - mmax[d]) / denom[d];
  atomicAdd(&outf[(size_t)d * OUTF + c], alpha * z[(size_t)s * OUTF + c]);
}

// gated attention score per row: A[r] = (tanh(row@Wa+ba) * sigm(row@Wb+bb)) @ Wc + bc
__global__ void attn_gate_kernel(const float* __restrict__ outf,
                                 const float* __restrict__ Wa, const float* __restrict__ ba,
                                 const float* __restrict__ Wb, const float* __restrict__ bb,
                                 const float* __restrict__ Wc, const float* __restrict__ bc,
                                 float* __restrict__ Ascore) {
  __shared__ float row[OUTF];
  __shared__ float partial[DATT];
  int r = blockIdx.x;
  int j = threadIdx.x;  // 64 threads
  for (int k = j; k < OUTF; k += DATT) row[k] = outf[(size_t)r * OUTF + k];
  __syncthreads();
  float sa = ba[j], sb = bb[j];
#pragma unroll 4
  for (int k = 0; k < OUTF; ++k) {
    float v = row[k];
    sa += v * Wa[(size_t)k * DATT + j];
    sb += v * Wb[(size_t)k * DATT + j];
  }
  float sig = 1.0f / (1.0f + __expf(-sb));
  partial[j] = tanhf(sa) * sig * Wc[j];
  __syncthreads();
  if (j == 0) {
    float ssum = bc[0];
#pragma unroll
    for (int t = 0; t < DATT; ++t) ssum += partial[t];
    Ascore[r] = ssum;
  }
}

// single-block softmax statistics over Ascore[n] -> red[0]=max, red[1]=sum(exp)
__global__ void softmax_stats_kernel(const float* __restrict__ A,
                                     float* __restrict__ red, int n) {
  __shared__ float sm[1024];
  int t = threadIdx.x;
  float mx = -3.4028235e38f;
  for (int i = t; i < n; i += 1024) mx = fmaxf(mx, A[i]);
  sm[t] = mx; __syncthreads();
  for (int s = 512; s > 0; s >>= 1) { if (t < s) sm[t] = fmaxf(sm[t], sm[t + s]); __syncthreads(); }
  float m = sm[0];
  __syncthreads();
  float sum = 0.0f;
  for (int i = t; i < n; i += 1024) sum += __expf(A[i] - m);
  sm[t] = sum; __syncthreads();
  for (int s = 512; s > 0; s >>= 1) { if (t < s) sm[t] += sm[t + s]; __syncthreads(); }
  if (t == 0) { red[0] = m; red[1] = sm[0]; }
}

// pooled[c] = sum_i softmax(A)_i * out[i,c]   (one block of 128 threads)
__global__ void pool_kernel(const float* __restrict__ A,
                            const float* __restrict__ red,
                            const float* __restrict__ outf,
                            float* __restrict__ dout, int n) {
  int c = threadIdx.x;
  float m = red[0], Z = red[1];
  float acc = 0.0f;
  for (int i = 0; i < n; ++i)
    acc += __expf(A[i] - m) * outf[(size_t)i * OUTF + c];
  dout[c] = acc / Z;
}

// ---------------------------------------------------------------------------
// Orchestration
// ---------------------------------------------------------------------------
extern "C" void kernel_launch(void* const* d_in, const int* in_sizes, int n_in,
                              void* d_out, int out_size, void* d_ws, size_t ws_size,
                              hipStream_t stream) {
  const float* x    = (const float*)d_in[0];   // [8192,512]
  const float* G    = (const float*)d_in[1];   // [8192,8192]
  const int*   src  = (const int*)  d_in[2];   // [E]
  const int*   dst  = (const int*)  d_in[3];   // [E]
  const float* W1   = (const float*)d_in[4];   // [512,256]
  const float* b1   = (const float*)d_in[5];   // [256]
  const float* W2   = (const float*)d_in[6];   // [256,128]
  const float* b2   = (const float*)d_in[7];   // [128]
  const float* Wg   = (const float*)d_in[8];   // [128,128]
  const float* a_s  = (const float*)d_in[9];   // [128]
  const float* a_d  = (const float*)d_in[10];  // [128]
  const float* Wa   = (const float*)d_in[11];  // [128,64]
  const float* ba   = (const float*)d_in[12];  // [64]
  const float* Wb   = (const float*)d_in[13];  // [128,64]
  const float* bb   = (const float*)d_in[14];  // [64]
  const float* Wc   = (const float*)d_in[15];  // [64,1]
  const float* bc   = (const float*)d_in[16];  // [1]
  float* out = (float*)d_out;                  // [128]

  const int n = NNODES;
  const int E = in_sizes[2];
  const int total_edges = E + n;

  // ---- workspace carve-out (256B aligned) ----
  uint8_t* wp = (uint8_t*)d_ws;
  auto carve = [&](size_t bytes) -> void* {
    void* r = (void*)wp;
    wp += (bytes + 255) & ~(size_t)255;
    return r;
  };
  bf16*  W1T   = (bf16*) carve((size_t)HIDF * FIN  * 2);  // [256][512]
  bf16*  W2T   = (bf16*) carve((size_t)OUTF * HIDF * 2);  // [128][256]
  bf16*  WgT   = (bf16*) carve((size_t)OUTF * OUTF * 2);  // [128][128]
  bf16*  T1T   = (bf16*) carve((size_t)HIDF * n    * 2);  // [256][8192] = (xW1+b1)^T
  bf16*  H1b   = (bf16*) carve((size_t)n * HIDF    * 2);  // relu(G@T1)  [8192,256]
  bf16*  T2T   = (bf16*) carve((size_t)OUTF * n    * 2);  // (H1 W2 + b2)^T [128][8192]
  bf16*  H2b   = (bf16*) carve((size_t)n * OUTF    * 2);  // G@T2  [8192,128]
  float* zf    = (float*)carve((size_t)n * OUTF    * 4);  // z = H2@Wg  [8192,128]
  float* zs    = (float*)carve((size_t)n * 4);
  float* zd    = (float*)carve((size_t)n * 4);
  float* mmax  = (float*)carve((size_t)n * 4);
  float* denom = (float*)carve((size_t)n * 4);
  float* outf  = (float*)carve((size_t)n * OUTF * 4);     // GAT output [8192,128]
  float* Ascr  = (float*)carve((size_t)n * 4);
  float* red   = (float*)carve(256);

  // ---- weight conversion (transpose + bf16) ----
  convT_bf16_kernel<<<(FIN * HIDF + 255) / 256, 256, 0, stream>>>(W1, W1T, FIN, HIDF);
  convT_bf16_kernel<<<(HIDF * OUTF + 255) / 256, 256, 0, stream>>>(W2, W2T, HIDF, OUTF);
  convT_bf16_kernel<<<(OUTF * OUTF + 255) / 256, 256, 0, stream>>>(Wg, WgT, OUTF, OUTF);

  // ---- GEMM chain (all WMMA bf16, f32 accumulate) ----
  // T1^T = (x @ W1 + b1)^T            (A f32, bias, no relu, bf16-transposed out)
  launch_gemm<float, 2, true,  false>(x,   W1T, b1,      (void*)T1T, n, HIDF, FIN,  stream);
  // H1 = relu(G @ T1)                 (A f32, no bias, relu, bf16 row-major out)
  launch_gemm<float, 1, false, true >(G,   T1T, nullptr, (void*)H1b, n, HIDF, n,    stream);
  // T2^T = (H1 @ W2 + b2)^T           (A bf16, bias, no relu, bf16-transposed out)
  launch_gemm<bf16,  2, true,  false>(H1b, W2T, b2,      (void*)T2T, n, OUTF, HIDF, stream);
  // H2 = G @ T2                       (A f32, no bias, no relu, bf16 row-major)
  launch_gemm<float, 1, false, false>(G,   T2T, nullptr, (void*)H2b, n, OUTF, n,    stream);
  // z = H2 @ Wg                       (A bf16, f32 row-major out)
  launch_gemm<bf16,  0, false, false>(H2b, WgT, nullptr, (void*)zf,  n, OUTF, OUTF, stream);

  // ---- GAT edge softmax + scatter ----
  vecmat_kernel<<<(n + 255) / 256, 256, 0, stream>>>(zf, a_s, a_d, zs, zd, n);
  fill_f32_kernel<<<(n + 255) / 256, 256, 0, stream>>>(mmax, -3.4028235e38f, n);
  fill_f32_kernel<<<(n + 255) / 256, 256, 0, stream>>>(denom, 0.0f, n);
  fill_f32_kernel<<<(n * OUTF + 255) / 256, 256, 0, stream>>>(outf, 0.0f, n * OUTF);
  edge_max_kernel<<<(total_edges + 255) / 256, 256, 0, stream>>>(src, dst, zs, zd, mmax, E, n);
  edge_sum_kernel<<<(total_edges + 255) / 256, 256, 0, stream>>>(src, dst, zs, zd, mmax, denom, E, n);
  edge_scatter_kernel<<<total_edges, OUTF, 0, stream>>>(src, dst, zs, zd, mmax, denom, zf, outf, E, n);
  relu_inplace_kernel<<<(n * OUTF + 255) / 256, 256, 0, stream>>>(outf, n * OUTF);

  // ---- gated attention pooling ----
  attn_gate_kernel<<<n, DATT, 0, stream>>>(outf, Wa, ba, Wb, bb, Wc, bc, Ascr);
  softmax_stats_kernel<<<1, 1024, 0, stream>>>(Ascr, red, n);
  pool_kernel<<<1, OUTF, 0, stream>>>(Ascr, red, outf, out, n);
}
